// DecoderWithAtt_40673340293552
// MI455X (gfx1250) — compile-verified, matrix operand order checked by reading, not run on hardware
//
#include <hip/hip_runtime.h>
#include <hip/hip_bf16.h>
#include <stdint.h>

#define H   512
#define V   32000
#define T1  128
#define T2  64
#define B   32

typedef __attribute__((ext_vector_type(16))) __bf16 bf16x16;
typedef __attribute__((ext_vector_type(8)))  float  f32x8;
typedef __attribute__((ext_vector_type(4)))  int    v4i;

#define GLB_AS __attribute__((address_space(1)))
#define LDS_AS __attribute__((address_space(3)))

#if defined(__HIP_DEVICE_COMPILE__) && __has_builtin(__builtin_amdgcn_global_load_async_to_lds_b128)
#define USE_ASYNC_LDS 1
#else
#define USE_ASYNC_LDS 0
#endif

__device__ __forceinline__ void wait_async0() {
#if __has_builtin(__builtin_amdgcn_s_wait_asynccnt)
    __builtin_amdgcn_s_wait_asynccnt(0);
#else
    asm volatile("s_wait_asynccnt 0x0" ::: "memory");
#endif
}

__device__ __forceinline__ __bf16 f2bf(float f) {
    unsigned u = __builtin_bit_cast(unsigned, f);
    unsigned short h = (unsigned short)((u + 0x7FFFu + ((u >> 16) & 1u)) >> 16);
    return __builtin_bit_cast(__bf16, h);
}
__device__ __forceinline__ float sigm(float x) { return 1.0f / (1.0f + expf(-x)); }

// ---------------------------------------------------------------------------
// Generic bf16 GEMM:  C[M,N] = A[M,K] * W[N,K]^T + bias[N]
// Block = 256 threads (8 waves). Macro tile 32(M) x 128(N), K chunk 32.
// Wave w owns N-tile w (16 cols) and both M-tiles (rows 0..15, 16..31).
// Staging: CDNA5 GLOBAL_LOAD_ASYNC_TO_LDS_B128 (ASYNCcnt), no VGPR round-trip.
// Fragment layouts per CDNA5 ISA 7.12.2 (16-bit A/B, wave32).
// ---------------------------------------------------------------------------
__global__ void __launch_bounds__(256)
k_gemm_bf16(const __bf16* __restrict__ A, const __bf16* __restrict__ W,
            const float* __restrict__ bias, float* __restrict__ C,
            int M, int N, int K) {
    __shared__ alignas(16) __bf16 sA[32 * 32];     // 2 KB
    __shared__ alignas(16) __bf16 sW[128 * 32];    // 8 KB

    const int tid  = threadIdx.x;
    const int wave = tid >> 5;
    const int lane = tid & 31;
    const int bn   = blockIdx.x;          // N / 128
    const int bm   = blockIdx.y;          // M / 32
    const int m0   = bm * 32;
    const int n0   = bn * 128 + wave * 16;

    const int lr = lane & 15;             // row within 16-wide tile (A:M, B:N)
    const int hi = (lane >> 4) & 1;       // lane half selects K sub-block

    f32x8 c0 = {};
    f32x8 c1 = {};

    for (int k0 = 0; k0 < K; k0 += 32) {
#if USE_ASYNC_LDS
        // ---- async LDS staging: 16 bytes (8 bf16) per transfer ----
        // A chunk: 32 rows x 64B = 128 units, waves 0..3 (uniform per wave)
        if (tid < 128) {
            int r = tid >> 2, c = tid & 3;
            const __bf16* g = A + (size_t)(m0 + r) * K + k0 + c * 8;
            __bf16* l = (__bf16*)((char*)sA + tid * 16);
            __builtin_amdgcn_global_load_async_to_lds_b128(
                (GLB_AS v4i*)g, (LDS_AS v4i*)l, 0, 0);
        }
        // W chunk: 128 rows x 64B = 512 units, 2 per thread
        #pragma unroll
        for (int u = tid; u < 512; u += 256) {
            int r = u >> 2, c = u & 3;
            const __bf16* g = W + (size_t)(bn * 128 + r) * K + k0 + c * 8;
            __bf16* l = (__bf16*)((char*)sW + u * 16);
            __builtin_amdgcn_global_load_async_to_lds_b128(
                (GLB_AS v4i*)g, (LDS_AS v4i*)l, 0, 0);
        }
        wait_async0();
#else
        const uint32_t* gA = (const uint32_t*)(A + (size_t)m0 * K + k0);
        const uint32_t* gW = (const uint32_t*)(W + (size_t)bn * 128 * K + k0);
        for (int i = tid; i < 512; i += 256) {
            int r = i >> 4, cw = i & 15;
            ((uint32_t*)sA)[r * 16 + cw] = gA[(size_t)r * (K >> 1) + cw];
        }
        for (int i = tid; i < 2048; i += 256) {
            int r = i >> 4, cw = i & 15;
            ((uint32_t*)sW)[r * 16 + cw] = gW[(size_t)r * (K >> 1) + cw];
        }
#endif
        if (k0 + 32 < K) {            // CDNA5 global_prefetch for next chunk
            __builtin_prefetch((const void*)(A + (size_t)m0 * K + k0 + 32), 0, 1);
            __builtin_prefetch((const void*)(W + (size_t)bn * 128 * K + k0 + 32), 0, 1);
        }
        __syncthreads();

        // build fragments (ISA 7.12.2: 16-bit A 16x32 / B 32x16 layouts);
        // per lane these are 2 contiguous 16B runs -> ds_load_b128 pairs
        bf16x16 a0, a1, bf;
        #pragma unroll
        for (int j = 0; j < 16; ++j) {
            const int v = j >> 1, p = j & 1;
            const int k = ((v >> 2) << 4) + (hi << 3) + ((v & 3) << 1) + p;
            a0[j] = sA[lr * 32 + k];
            a1[j] = sA[(16 + lr) * 32 + k];
            bf[j] = sW[(wave * 16 + lr) * 32 + k];
        }
        c0 = __builtin_amdgcn_wmma_f32_16x16x32_bf16(false, a0, false, bf, (short)0, c0, false, false);
        c1 = __builtin_amdgcn_wmma_f32_16x16x32_bf16(false, a1, false, bf, (short)0, c1, false, false);
        __syncthreads();
    }

    const int col = n0 + lr;
    const float bv = bias ? bias[col] : 0.0f;
    #pragma unroll
    for (int r = 0; r < 8; ++r) {
        C[(size_t)(m0 + r + hi * 8) * N + col]      = c0[r] + bv;
        C[(size_t)(m0 + 16 + r + hi * 8) * N + col] = c1[r] + bv;
    }
}

// ---------------------------------------------------------------------------
// Elementwise / helper kernels
// ---------------------------------------------------------------------------
__global__ void k_convert_bf16(const float* __restrict__ s, __bf16* __restrict__ d, int n) {
    int i = blockIdx.x * 256 + threadIdx.x;
    if (i < n) d[i] = f2bf(s[i]);
}

// split wa_W[H, 2H] -> Wx[H,H] (cols 0..H-1) and We[H,H] (cols H..2H-1), bf16
__global__ void k_split_waW(const float* __restrict__ waW,
                            __bf16* __restrict__ Wx, __bf16* __restrict__ We) {
    int i = blockIdx.x * 256 + threadIdx.x;   // H*2H
    int r = i / (2 * H), c = i % (2 * H);
    __bf16 v = f2bf(waW[i]);
    if (c < H) Wx[r * H + c] = v;
    else       We[r * H + (c - H)] = v;
}

// embedding gather: x_bf[row, :] = bf16(emb[ids[row], :]),  row in [0, T2*B)
__global__ void k_embed(const int* __restrict__ ids, const float* __restrict__ emb,
                        __bf16* __restrict__ xb) {
    int row = blockIdx.x;
    int id  = ids[row];
    const float* src = emb + (size_t)id * H;
    __bf16* dst = xb + (size_t)row * H;
    for (int h = threadIdx.x; h < H; h += blockDim.x) dst[h] = f2bf(src[h]);
}

__global__ void k_init_h(const float* __restrict__ state,
                         float* h0, float* h1, __bf16* h0b, __bf16* h1b) {
    int i = blockIdx.x * 256 + threadIdx.x;   // B*H
    float a = state[i], c = state[B * H + i];
    h0[i] = a; h1[i] = c; h0b[i] = f2bf(a); h1b[i] = f2bf(c);
}

// e[t1,b] = sum_h tanh(qt[b,h] + enc_proj[t1,b,h]) * va[h]    (one wave per (t1,b))
__global__ void __launch_bounds__(256)
k_e(const float* __restrict__ qt, const float* __restrict__ enc_proj,
    const float* __restrict__ va, float* __restrict__ e) {
    int gw   = blockIdx.x * 8 + (threadIdx.x >> 5);
    int lane = threadIdx.x & 31;
    int t1 = gw / B, b = gw % B;
    const float* ep = enc_proj + ((size_t)t1 * B + b) * H;
    const float* q  = qt + (size_t)b * H;
    float acc = 0.f;
    for (int h = lane; h < H; h += 32) acc += tanhf(q[h] + ep[h]) * va[h];
    #pragma unroll
    for (int m = 16; m > 0; m >>= 1) acc += __shfl_xor(acc, m, 32);
    if (lane == 0) e[t1 * B + b] = acc;
}

// per-batch softmax over T1, context vector, bf16 att half of gin; last_score out
__global__ void __launch_bounds__(128)
k_softmax_att(const float* __restrict__ e, const float* __restrict__ enc,
              float* __restrict__ att_v, __bf16* __restrict__ gin_bf,
              float* __restrict__ last_out) {
    __shared__ float sc[T1];
    __shared__ float red[T1];
    const int b = blockIdx.x, tid = threadIdx.x;
    float v = e[tid * B + b];
    red[tid] = v; __syncthreads();
    for (int s = 64; s > 0; s >>= 1) { if (tid < s) red[tid] = fmaxf(red[tid], red[tid + s]); __syncthreads(); }
    float mx = red[0]; __syncthreads();
    float p = expf(v - mx);
    red[tid] = p; __syncthreads();
    for (int s = 64; s > 0; s >>= 1) { if (tid < s) red[tid] += red[tid + s]; __syncthreads(); }
    float sv = p / red[0];
    sc[tid] = sv;
    if (last_out) last_out[tid * B + b] = sv;
    __syncthreads();
    for (int h = tid; h < H; h += 128) {
        float acc = 0.f;
        for (int t = 0; t < T1; ++t) acc += sc[t] * enc[((size_t)t * B + b) * H + h];
        att_v[b * H + h] = acc;
        gin_bf[b * 2 * H + H + h] = f2bf(acc);   // gin = [xt | att_v]
    }
}

// gin[:, :H] = xt  (bf16)
__global__ void k_gin_x(const __bf16* __restrict__ xb_t, __bf16* __restrict__ gin) {
    int i = blockIdx.x * 256 + threadIdx.x;   // B*H
    int b = i / H, h = i % H;
    gin[b * 2 * H + h] = xb_t[i];
}

// fused GRU gate math; h updated in place; bf16 shadow; optional ys row copy
__global__ void __launch_bounds__(256)
k_gru(const float* __restrict__ gi, const float* __restrict__ gh,
      float* __restrict__ h, __bf16* __restrict__ hb, __bf16* __restrict__ ys) {
    int i = blockIdx.x * 256 + threadIdx.x;   // B*H
    int b = i / H, hh = i % H;
    float ir = gi[b * 3 * H + hh], iz = gi[b * 3 * H + H + hh], in = gi[b * 3 * H + 2 * H + hh];
    float hr = gh[b * 3 * H + hh], hz = gh[b * 3 * H + H + hh], hn = gh[b * 3 * H + 2 * H + hh];
    float r = sigm(ir + hr);
    float z = sigm(iz + hz);
    float n = tanhf(in + r * hn);
    float out = (1.f - z) * n + z * h[i];
    h[i]  = out;
    hb[i] = f2bf(out);
    if (ys) ys[i] = f2bf(out);
}

__global__ void k_finalize(const float* __restrict__ h0, const float* __restrict__ h1,
                           float* __restrict__ out_h) {
    int i = blockIdx.x * 256 + threadIdx.x;   // 2*B*H
    out_h[i] = (i < B * H) ? h0[i] : h1[i - B * H];
}

// ---------------------------------------------------------------------------
extern "C" void kernel_launch(void* const* d_in, const int* in_sizes, int n_in,
                              void* d_out, int out_size, void* d_ws, size_t ws_size,
                              hipStream_t stream) {
    const int*   ids   = (const int*)  d_in[0];
    const float* state = (const float*)d_in[1];
    const float* enc   = (const float*)d_in[2];
    const float* emb   = (const float*)d_in[3];
    const float* waW   = (const float*)d_in[4];
    const float* wab   = (const float*)d_in[5];
    const float* vaW   = (const float*)d_in[6];
    const float* Wih0  = (const float*)d_in[7];
    const float* Whh0  = (const float*)d_in[8];
    const float* bih0  = (const float*)d_in[9];
    const float* bhh0  = (const float*)d_in[10];
    const float* Wih1  = (const float*)d_in[11];
    const float* Whh1  = (const float*)d_in[12];
    const float* bih1  = (const float*)d_in[13];
    const float* bhh1  = (const float*)d_in[14];
    const float* outW  = (const float*)d_in[15];
    const float* outb  = (const float*)d_in[16];
    float* out = (float*)d_out;

    // deterministic workspace carve
    char* ws = (char*)d_ws;
    size_t off = 0;
    auto alloc = [&](size_t bytes) -> void* {
        off = (off + 255) & ~(size_t)255;
        void* p = ws + off;
        off += bytes;
        return p;
    };
    __bf16* x_bf    = (__bf16*)alloc((size_t)T2 * B * H * 2);
    __bf16* enc_bf  = (__bf16*)alloc((size_t)T1 * B * H * 2);
    __bf16* Wx_bf   = (__bf16*)alloc((size_t)H * H * 2);
    __bf16* We_bf   = (__bf16*)alloc((size_t)H * H * 2);
    __bf16* Wih0_bf = (__bf16*)alloc((size_t)3 * H * 2 * H * 2);
    __bf16* Whh0_bf = (__bf16*)alloc((size_t)3 * H * H * 2);
    __bf16* Wih1_bf = (__bf16*)alloc((size_t)3 * H * H * 2);
    __bf16* Whh1_bf = (__bf16*)alloc((size_t)3 * H * H * 2);
    __bf16* outW_bf = (__bf16*)alloc((size_t)V * H * 2);
    __bf16* ys_bf   = (__bf16*)alloc((size_t)T2 * B * H * 2);
    float*  enc_pj  = (float*)alloc((size_t)T1 * B * H * 4);
    float*  qt      = (float*)alloc((size_t)B * H * 4);
    float*  ebuf    = (float*)alloc((size_t)T1 * B * 4);
    float*  att_v   = (float*)alloc((size_t)B * H * 4);
    __bf16* gin_bf  = (__bf16*)alloc((size_t)B * 2 * H * 2);
    float*  gi0     = (float*)alloc((size_t)B * 3 * H * 4);
    float*  gh0     = (float*)alloc((size_t)B * 3 * H * 4);
    float*  gi1     = (float*)alloc((size_t)B * 3 * H * 4);
    float*  gh1     = (float*)alloc((size_t)B * 3 * H * 4);
    float*  h0      = (float*)alloc((size_t)B * H * 4);
    float*  h1      = (float*)alloc((size_t)B * H * 4);
    __bf16* h0b     = (__bf16*)alloc((size_t)B * H * 2);
    __bf16* h1b     = (__bf16*)alloc((size_t)B * H * 2);
    (void)ws_size; (void)in_sizes; (void)n_in; (void)out_size;

    auto gemm = [&](const __bf16* A, const __bf16* Wm, const float* bias, float* C,
                    int M, int N, int K) {
        dim3 g(N / 128, M / 32);
        k_gemm_bf16<<<g, 256, 0, stream>>>(A, Wm, bias, C, M, N, K);
    };
    auto conv = [&](const float* s, __bf16* d, int n) {
        k_convert_bf16<<<(n + 255) / 256, 256, 0, stream>>>(s, d, n);
    };

    // ---- one-time prep ----
    k_embed<<<T2 * B, 256, 0, stream>>>(ids, emb, x_bf);
    conv(enc,  enc_bf,  T1 * B * H);
    k_split_waW<<<(H * 2 * H) / 256, 256, 0, stream>>>(waW, Wx_bf, We_bf);
    conv(Wih0, Wih0_bf, 3 * H * 2 * H);
    conv(Whh0, Whh0_bf, 3 * H * H);
    conv(Wih1, Wih1_bf, 3 * H * H);
    conv(Whh1, Whh1_bf, 3 * H * H);
    conv(outW, outW_bf, V * H);
    k_init_h<<<(B * H) / 256, 256, 0, stream>>>(state, h0, h1, h0b, h1b);

    // enc_proj[t1*B+b, o] = enc @ We^T + wa_b
    gemm(enc_bf, We_bf, wab, enc_pj, T1 * B, H, H);

    const size_t y_elems = (size_t)T2 * B * V;
    float* out_hfinal = out + y_elems;
    float* out_lastsc = out + y_elems + 2 * B * H;

    // ---- sequential decoder steps ----
    for (int t = 0; t < T2; ++t) {
        const __bf16* xt = x_bf + (size_t)t * B * H;
        // q = xt @ Wx^T   (bias folded into enc_proj)
        gemm(xt, Wx_bf, nullptr, qt, B, H, H);
        k_e<<<(T1 * B) / 8, 256, 0, stream>>>(qt, enc_pj, vaW, ebuf);
        k_softmax_att<<<B, 128, 0, stream>>>(ebuf, enc, att_v, gin_bf,
                                             (t == T2 - 1) ? out_lastsc : nullptr);
        k_gin_x<<<(B * H) / 256, 256, 0, stream>>>(xt, gin_bf);
        // layer 0
        gemm(gin_bf, Wih0_bf, bih0, gi0, B, 3 * H, 2 * H);
        gemm(h0b,    Whh0_bf, bhh0, gh0, B, 3 * H, H);
        k_gru<<<(B * H) / 256, 256, 0, stream>>>(gi0, gh0, h0, h0b, nullptr);
        // layer 1 (gh1 reads old h1b before k_gru overwrites it)
        gemm(h0b, Wih1_bf, bih1, gi1, B, 3 * H, H);
        gemm(h1b, Whh1_bf, bhh1, gh1, B, 3 * H, H);
        k_gru<<<(B * H) / 256, 256, 0, stream>>>(gi1, gh1, h1, h1b,
                                                 ys_bf + (size_t)t * B * H);
    }

    // ---- vocab projection: the 67-GFLOP GEMM (HBM-store-bound) ----
    gemm(ys_bf, outW_bf, outb, out, T2 * B, V, H);

    // ---- h_final ----
    k_finalize<<<(2 * B * H) / 256, 256, 0, stream>>>(h0, h1, out_hfinal);
}
